// GraphAttentionLayer_3204045602881
// MI455X (gfx1250) — compile-verified, matrix operand order checked by reading, not run on hardware
//
#include <hip/hip_runtime.h>
#include <hip/hip_bf16.h>

// ---------------------------------------------------------------------------
// GAT layer for MI455X (gfx1250, wave32, WMMA).
//
// Roofline: adj read (256 MB) bounds runtime at ~11us @ 23.3 TB/s.
// attention@h = 17.2 GFLOP -> v_wmma_f32_16x16x32_f16 (f16 in, f32 acc)
// keeps math far below the memory floor. hT (2 MB f16) stays L2-resident.
// Fixed softmax shift M_i = leaky(f1_i + max_j f2_j) (valid upper bound,
// leaky_relu monotone) removes online-softmax rescaling -> single pass.
// Column tile = 64 -> 2 WMMAs per wave per barrier (halved sync overhead).
// ---------------------------------------------------------------------------

#define GAT_N       8192
#define GAT_IN      256
#define GAT_OUT     128
#define GAT_ALPHA   0.2f

typedef _Float16 v16h __attribute__((ext_vector_type(16)));
typedef _Float16 v8h  __attribute__((ext_vector_type(8)));
typedef _Float16 v4h  __attribute__((ext_vector_type(4)));
typedef _Float16 v2h  __attribute__((ext_vector_type(2)));
typedef float    v8f  __attribute__((ext_vector_type(8)));

#define SHUF16(lo, hi) __builtin_shufflevector(lo, hi, 0,1,2,3,4,5,6,7,8,9,10,11,12,13,14,15)

// gemm_h LDS tile: 32 halves/row, stride 40 (80 B = 5x16 B aligned rows).
#define XSTRIDE 40
// gat_main LDS p-tile: 64 halves/row, stride 72 (144 B = 9x16 B aligned rows;
// 36 dwords/row, gcd(36,64)=4 but 9 coprime 16 -> producer stores conflict-free).
#define PSTRIDE 72

// --------------------------------------------------------------------------
// Kernel 0: WT[c][k] = f16(W[k][c])   (column-major f16 copy of W, 64 KB)
// --------------------------------------------------------------------------
__global__ void gat_prep_wt(const float* __restrict__ W, _Float16* __restrict__ WT) {
  int c = blockIdx.x;        // 0..127
  int k = threadIdx.x;       // 0..255
  WT[(size_t)c * GAT_IN + k] = (_Float16)W[(size_t)k * GAT_OUT + c];
}

// --------------------------------------------------------------------------
// Kernel 1: h = x @ W via WMMA.  Block = 256 thr (8 waves) -> 16 rows x 128 cols.
// Wave w owns the 16x16 tile at columns [16w, 16w+16).  K = 256 in 8 steps of 32.
// Writes h (f32, row major, for f1/f2) and hT (f16, [col][row], B operand later).
// --------------------------------------------------------------------------
__global__ void gat_gemm_h(const float* __restrict__ x, const _Float16* __restrict__ WT,
                           float* __restrict__ h, _Float16* __restrict__ hT) {
  __shared__ __align__(16) _Float16 a_lds[16 * XSTRIDE];

  const int t    = threadIdx.x;
  const int wid  = t >> 5;
  const int lane = t & 31;
  const int lo   = lane & 15;          // C: N / A: M
  const int hi   = lane >> 4;
  const int kb   = hi * 8;             // K sub-group base (A/B f16 layout)
  const int m    = t & 15;             // producer row
  const int ks   = (t >> 4) * 2;       // producer K slot (2 halves)
  const int i0   = blockIdx.x * 16;

  const _Float16* bbase = WT + (size_t)(16 * wid + lo) * GAT_IN + kb;

  v8f acc = {};
  for (int kt = 0; kt < 8; ++kt) {
    const int k0 = kt * 32;
    // stage x tile (16x32) into LDS as f16, row-major
    float2 xv = *(const float2*)(x + (size_t)(i0 + m) * GAT_IN + k0 + ks);
    v2h pv; pv[0] = (_Float16)xv.x; pv[1] = (_Float16)xv.y;
    *(v2h*)&a_lds[m * XSTRIDE + ks] = pv;
    __syncthreads();

    // A (16x32 f16): lane lo = row M; K = {kb..kb+7} and {kb+16..kb+23}
    v8h alo = *(const v8h*)&a_lds[lo * XSTRIDE + kb];
    v8h ahi = *(const v8h*)&a_lds[lo * XSTRIDE + kb + 16];
    v16h A = SHUF16(alo, ahi);
    // B (32x16 f16) from column-major WT: lane lo = col N, same K grouping
    v8h blo = *(const v8h*)(bbase + k0);
    v8h bhi = *(const v8h*)(bbase + k0 + 16);
    v16h B = SHUF16(blo, bhi);

    acc = __builtin_amdgcn_wmma_f32_16x16x32_f16(false, A, false, B,
                                                 (short)0, acc, false, false);
    __syncthreads();
  }

  // C layout: VGPR r, row M = r + 8*hi, col N = lo
  const int c = 16 * wid + lo;
#pragma unroll
  for (int r = 0; r < 8; ++r) {
    const int row = i0 + r + 8 * hi;
    const float v = acc[r];
    h[(size_t)row * GAT_OUT + c]   = v;               // coalesced
    hT[(size_t)c * GAT_N + row]    = (_Float16)v;     // scattered b16 (2 MB total, one-time)
  }
}

// --------------------------------------------------------------------------
// Kernel 2: f1[i] = h[i,:]@a1, f2[i] = h[i,:]@a2.  One wave32 per row.
// --------------------------------------------------------------------------
__global__ void gat_f1f2(const float* __restrict__ h, const float* __restrict__ a,
                         float* __restrict__ f1, float* __restrict__ f2) {
  const int wid  = threadIdx.x >> 5;
  const int lane = threadIdx.x & 31;
  const int row  = blockIdx.x * 8 + wid;

  float4 hv  = *(const float4*)(h + (size_t)row * GAT_OUT + lane * 4);
  float4 a1v = *(const float4*)(a + lane * 4);
  float4 a2v = *(const float4*)(a + GAT_OUT + lane * 4);
  float s1 = hv.x * a1v.x + hv.y * a1v.y + hv.z * a1v.z + hv.w * a1v.w;
  float s2 = hv.x * a2v.x + hv.y * a2v.y + hv.z * a2v.z + hv.w * a2v.w;
#pragma unroll
  for (int o = 16; o > 0; o >>= 1) {
    s1 += __shfl_xor(s1, o, 32);
    s2 += __shfl_xor(s2, o, 32);
  }
  if (lane == 0) { f1[row] = s1; f2[row] = s2; }
}

// --------------------------------------------------------------------------
// Kernel 3: F2max = max_j f2[j]   (single 1024-thread block)
// --------------------------------------------------------------------------
__global__ void gat_f2max(const float* __restrict__ f2, float* __restrict__ f2max) {
  __shared__ float red[1024];
  const int t = threadIdx.x;
  float m = -3.4e38f;
  for (int i = t; i < GAT_N; i += 1024) m = fmaxf(m, f2[i]);
  red[t] = m;
  __syncthreads();
  for (int s = 512; s > 0; s >>= 1) {
    if (t < s) red[t] = fmaxf(red[t], red[t + s]);
    __syncthreads();
  }
  if (t == 0) f2max[0] = red[0];
}

// --------------------------------------------------------------------------
// Kernel 4: fused masked softmax + (P @ h) + ELU.
// Block = 256 thr (8 waves) owns 16 output rows; loops over 8192 cols in
// tiles of 64.  Per tile: each thread produces 4 probabilities (int4 adj +
// float4 f2 loads, one v4h LDS store); after one barrier each wave performs
// 4x ds_load_b128 (A), 4x global_load_b128 (B from L2-resident hT), and
// 2x v_wmma_f32_16x16x32_f16.  p-tile double buffered -> 1 barrier / tile.
// --------------------------------------------------------------------------
__global__ void gat_main(const int* __restrict__ adj, const _Float16* __restrict__ hT,
                         const float* __restrict__ f1, const float* __restrict__ f2,
                         const float* __restrict__ f2max, float* __restrict__ out) {
  __shared__ __align__(16) _Float16 p_lds[2][16 * PSTRIDE];
  __shared__ float l_lds[16];

  const int t    = threadIdx.x;
  const int wid  = t >> 5;
  const int lane = t & 31;
  const int lo   = lane & 15;
  const int hi   = lane >> 4;
  const int kb   = hi * 8;
  const int m    = t & 15;             // producer row
  const int ks   = (t >> 4) * 4;       // producer K slot (4 halves), 16 groups x 4 = 64
  const int i0   = blockIdx.x * 16;

  if (t < 16) l_lds[t] = 0.0f;
  __syncthreads();

  // fixed per-row softmax shift: leaky_relu(f1_i + max_j f2_j) >= all scores
  const float F2M   = f2max[0];
  const float f1m   = f1[i0 + m];
  const float zsh   = f1m + F2M;
  const float shift = fmaxf(zsh, GAT_ALPHA * zsh);

  const int*      adjp  = adj + (size_t)(i0 + m) * GAT_N + ks;
  const _Float16* bbase = hT + (size_t)(16 * wid + lo) * GAT_N + kb;

  float lsum = 0.0f;
  v8f acc = {};

  for (int jt = 0; jt < GAT_N / 64; ++jt) {
    const int j0 = jt * 64;
    // ---- produce 4 probabilities: p = adj ? exp(leaky(f1+f2) - shift) : 0 ----
    int4   av  = *(const int4*)(adjp + j0);
    float4 f2v = *(const float4*)(f2 + j0 + ks);
    if (j0 + 256 < GAT_N) __builtin_prefetch(adjp + j0 + 256, 0, 0); // global_prefetch_b8

    float z0 = f1m + f2v.x, z1 = f1m + f2v.y, z2 = f1m + f2v.z, z3 = f1m + f2v.w;
    float e0 = fmaxf(z0, GAT_ALPHA * z0);
    float e1 = fmaxf(z1, GAT_ALPHA * z1);
    float e2 = fmaxf(z2, GAT_ALPHA * z2);
    float e3 = fmaxf(z3, GAT_ALPHA * z3);
    float p0 = (av.x > 0) ? __expf(e0 - shift) : 0.0f;
    float p1 = (av.y > 0) ? __expf(e1 - shift) : 0.0f;
    float p2 = (av.z > 0) ? __expf(e2 - shift) : 0.0f;
    float p3 = (av.w > 0) ? __expf(e3 - shift) : 0.0f;
    v4h pv;
    pv[0] = (_Float16)p0; pv[1] = (_Float16)p1;
    pv[2] = (_Float16)p2; pv[3] = (_Float16)p3;
    // denominator consistent with f16 A operand fed to the WMMA
    lsum += (float)pv[0] + (float)pv[1] + (float)pv[2] + (float)pv[3];
    const int buf = jt & 1;
    *(v4h*)&p_lds[buf][m * PSTRIDE + ks] = pv;
    __syncthreads();

    // ---- consume: two K=32 sub-tiles -> 2 WMMAs ----
    const _Float16* prow = &p_lds[buf][lo * PSTRIDE + kb];
    v8h a0lo = *(const v8h*)(prow);
    v8h a0hi = *(const v8h*)(prow + 16);
    v8h a1lo = *(const v8h*)(prow + 32);
    v8h a1hi = *(const v8h*)(prow + 48);
    v16h A0 = SHUF16(a0lo, a0hi);
    v16h A1 = SHUF16(a1lo, a1hi);

    const _Float16* brow = bbase + j0;
    v8h b0lo = *(const v8h*)(brow);
    v8h b0hi = *(const v8h*)(brow + 16);
    v8h b1lo = *(const v8h*)(brow + 32);
    v8h b1hi = *(const v8h*)(brow + 48);
    v16h B0 = SHUF16(b0lo, b0hi);
    v16h B1 = SHUF16(b1lo, b1hi);

    acc = __builtin_amdgcn_wmma_f32_16x16x32_f16(false, A0, false, B0,
                                                 (short)0, acc, false, false);
    acc = __builtin_amdgcn_wmma_f32_16x16x32_f16(false, A1, false, B1,
                                                 (short)0, acc, false, false);
    // no second barrier: reads of buf complete before the barrier preceding
    // the next write to the same buffer (2 iterations later)
  }

  atomicAdd(&l_lds[m], lsum);          // ds_add_f32
  __syncthreads();

  const int c = 16 * wid + lo;
#pragma unroll
  for (int r = 0; r < 8; ++r) {
    const int   row = r + 8 * hi;
    const float l   = l_lds[row];
    float v = acc[r] / l;              // h_prime
    v = (v > 0.0f) ? v : (__expf(v) - 1.0f);   // ELU (alpha=1)
    out[(size_t)(i0 + row) * GAT_OUT + c] = v;
  }
}

// --------------------------------------------------------------------------
extern "C" void kernel_launch(void* const* d_in, const int* in_sizes, int n_in,
                              void* d_out, int out_size, void* d_ws, size_t ws_size,
                              hipStream_t stream) {
  const float* x   = (const float*)d_in[0];   // [8192,256]
  const int*   adj = (const int*)  d_in[1];   // [8192,8192]
  const float* W   = (const float*)d_in[2];   // [256,128]
  const float* a   = (const float*)d_in[3];   // [256,1]
  float*       out = (float*)d_out;           // [8192,128]

  char* ws = (char*)d_ws;
  size_t off = 0;
  _Float16* hT = (_Float16*)(ws + off); off += (size_t)GAT_OUT * GAT_N * 2;   // 2 MB
  float*    h  = (float*)   (ws + off); off += (size_t)GAT_N * GAT_OUT * 4;   // 4 MB
  _Float16* WT = (_Float16*)(ws + off); off += (size_t)GAT_OUT * GAT_IN * 2;  // 64 KB
  float*    f1 = (float*)   (ws + off); off += (size_t)GAT_N * 4;
  float*    f2 = (float*)   (ws + off); off += (size_t)GAT_N * 4;
  float*    fm = (float*)   (ws + off); off += 256;

  gat_prep_wt<<<GAT_OUT, GAT_IN, 0, stream>>>(W, WT);
  gat_gemm_h <<<GAT_N / 16, 256, 0, stream>>>(x, WT, h, hT);
  gat_f1f2   <<<GAT_N / 8, 256, 0, stream>>>(h, a, f1, f2);
  gat_f2max  <<<1, 1024, 0, stream>>>(f2, fm);
  gat_main   <<<GAT_N / 16, 256, 0, stream>>>(adj, hT, f1, f2, fm, out);
}
